// OHDeltaProductSSM_62302795596519
// MI455X (gfx1250) — matrix operand order repaired and checked by reading, not compiled, not run on hardware
//
#include <hip/hip_runtime.h>
#include <hip/hip_bf16.h>

typedef __attribute__((ext_vector_type(16))) __bf16 v16bf;
typedef __attribute__((ext_vector_type(8)))  __bf16 v8bf_t;
typedef __attribute__((ext_vector_type(4)))  __bf16 v4bf_t;
typedef __attribute__((ext_vector_type(8)))  float  v8f;
typedef unsigned int u32x4 __attribute__((ext_vector_type(4)));
typedef int          i32x4 __attribute__((ext_vector_type(4)));
typedef int          i32x8 __attribute__((ext_vector_type(8)));

// ---------------- WMMA helpers ----------------

static __device__ __forceinline__ v8f wmma_bf16(v16bf a, v16bf b, v8f c) {
  // D = A(16x32 bf16) x B(32x16 bf16) + C(16x16 f32)
  return __builtin_amdgcn_wmma_f32_16x16x32_bf16(false, a, false, b, (short)0, c, false, false);
}

// A fragment: 16x32 bf16. Lane L<16 holds row M=L, K = {kh..kh+7, kh+16..kh+23}
// with kh = 0 for lanes 0-15 and kh = 8 for lanes 16-31 (ISA 7.12.2 table).
static __device__ __forceinline__ v16bf frag_a(const __bf16* tile, int stride, int rowBase, int lane) {
  const int rl = lane & 15;
  const int kh = (lane >> 4) << 3;           // 0 or 8
  const __bf16* p = tile + (rowBase + rl) * stride + kh;
  v8bf_t lo = *(const v8bf_t*)p;             // K = kh .. kh+7
  v8bf_t hi = *(const v8bf_t*)(p + 16);      // K = kh+16 .. kh+23
  v16bf r;
  #pragma unroll
  for (int i = 0; i < 8; ++i) { r[i] = lo[i]; r[i + 8] = hi[i]; }
  return r;
}

// B fragment: 32x16 bf16. Lane L<16 holds column N=L, K = {kh..kh+7, kh+16..kh+23}
// with kh = 0 for lanes 0-15 and kh = 8 for lanes 16-31 (matches frag_a).
static __device__ __forceinline__ v16bf frag_b(const __bf16* tile, int stride, int colBase, int lane) {
  const int cl = lane & 15;
  const int ks = (lane >> 4) << 3;           // 0 or 8
  const __bf16* p = tile + (colBase + cl) * stride + ks;
  v8bf_t lo = *(const v8bf_t*)p;
  v8bf_t hi = *(const v8bf_t*)(p + 16);
  v16bf r;
  #pragma unroll
  for (int i = 0; i < 8; ++i) { r[i] = lo[i]; r[i + 8] = hi[i]; }
  return r;
}

// ---------------- Tensor Data Mover: 2-D bf16 tile -> padded LDS ----------------
// D# per CDNA5 ISA. 2-D tile, data_size=2B, LDS padding
// via pad_interval/pad_amount so rows land at the bank-conflict-free stride the
// WMMA fragment loaders expect. Issued by one wave; tracked with TENSORcnt.

static __device__ __forceinline__ void tdm_load_2d_bf16(
    unsigned lds_addr, const __bf16* gptr,
    unsigned tile_x_elems, unsigned tile_rows,
    unsigned row_stride_elems,
    unsigned pad_interval_enc, unsigned pad_amount_enc) {
  unsigned long long ga = (unsigned long long)(size_t)gptr;
  u32x4 g0;
  g0[0] = 1u;                                            // count=1 (valid), user mode
  g0[1] = lds_addr;                                      // LDS byte address
  g0[2] = (unsigned)(ga & 0xFFFFFFFFull);                // global_addr[31:0]
  g0[3] = (unsigned)((ga >> 32) & 0x01FFFFFFull) | (2u << 30);  // addr[56:32] | type=2
  const unsigned td0 = row_stride_elems;                 // tensor_dim0 (row length)
  const unsigned td1 = 1u << 20;                         // generous tensor_dim1
  i32x8 g1;
  g1[0] = (int)((1u << 16)                               // data_size = 2 bytes
              | (1u << 20)                               // pad_enable
              | (pad_interval_enc << 22)
              | (pad_amount_enc << 25));
  g1[1] = (int)((td0 & 0xFFFFu) << 16);                  // atomic_barrier=0 | dim0.lo16
  g1[2] = (int)(((td0 >> 16) & 0xFFFFu) | ((td1 & 0xFFFFu) << 16));
  g1[3] = (int)(((td1 >> 16) & 0xFFFFu) | ((tile_x_elems & 0xFFFFu) << 16)); // tile_dim0
  g1[4] = (int)(tile_rows & 0xFFFFu);                    // tile_dim1 | tile_dim2=0
  g1[5] = (int)row_stride_elems;                         // tensor_dim0_stride lo32
  g1[6] = 0;
  g1[7] = 0;
  i32x4 z4; z4[0] = 0; z4[1] = 0; z4[2] = 0; z4[3] = 0;  // groups 2/3 unused (2-D)
#if defined(__clang_major__) && (__clang_major__ >= 23)
  i32x8 z8; 
  #pragma unroll
  for (int i = 0; i < 8; ++i) z8[i] = 0;
  __builtin_amdgcn_tensor_load_to_lds(g0, g1, z4, z4, z8, 0);
#else
  __builtin_amdgcn_tensor_load_to_lds(g0, g1, z4, z4, 0);
#endif
}

// ---------------- pack kernels ----------------

// Wc: 512 rows x 1024 (bf16). rows [0,64)=W_A, [64,320)=W_k flat, [320,322)=W_beta,
// [322,450)=B_w, [450,512)=0.
__global__ __launch_bounds__(256) void pack_w_kernel(
    const float* __restrict__ W_A, const float* __restrict__ W_k,
    const float* __restrict__ W_beta, const float* __restrict__ B_w,
    __bf16* __restrict__ Wc) {
  int idx = blockIdx.x * 256 + threadIdx.x;        // 512*1024 threads
  int r = idx >> 10, k = idx & 1023;
  float v;
  if (r < 64)        v = W_A[r * 1024 + k];
  else if (r < 320)  v = W_k[(r - 64) * 1024 + k];
  else if (r < 322)  v = W_beta[(r - 320) * 1024 + k];
  else if (r < 450)  v = B_w[(r - 322) * 1024 + k];
  else               v = 0.0f;
  Wc[idx] = (__bf16)v;
}

__global__ __launch_bounds__(256) void pack_c_kernel(const float* __restrict__ C, __bf16* __restrict__ Cb) {
  int idx = blockIdx.x * 256 + threadIdx.x;        // 1024*128 threads
  Cb[idx] = (__bf16)C[idx];
}

// ---------------- GEMM1: P[16384x512] = u[16384x1024] * Wc^T ----------------
// A-tile: f32 -> bf16 via VGPR staging (TDM cannot convert). B-tile: TDM DMA
// straight into padded LDS (stride 40: 32 data + 8 pad bf16 per row).

__global__ __launch_bounds__(256) void gemm1_kernel(
    const float* __restrict__ U, const __bf16* __restrict__ Wc, float* __restrict__ P) {
  __shared__ __bf16 As[2][128 * 40];
  __shared__ __bf16 Bs[2][128 * 40];
  const int tid = threadIdx.x, lane = tid & 31, w = tid >> 5;
  const int tileM = blockIdx.y * 128, tileN = blockIdx.x * 128;
  const int waveM = w & 3, waveN = w >> 2;         // 4x2 waves -> each 32 rows x 64 cols

  // stage 0: B via TDM (pad 16 dwords interval / 4 dwords amount -> stride 40)
  if (w == 0) {
    tdm_load_2d_bf16((unsigned)(size_t)&Bs[0][0], Wc + (size_t)tileN * 1024,
                     /*tile_x=*/32, /*rows=*/128, /*stride=*/1024,
                     /*pad_interval: 16 dw=*/3, /*pad_amount: 4 dw=*/3);
  }
  #pragma unroll
  for (int it = 0; it < 4; ++it) {                 // A stage 0: f32 -> bf16
    int v = tid + 256 * it; int row = v >> 3; int c4 = (v & 7) * 4;
    float4 f = *(const float4*)&U[(size_t)(tileM + row) * 1024 + c4];
    v4bf_t b; b[0] = (__bf16)f.x; b[1] = (__bf16)f.y; b[2] = (__bf16)f.z; b[3] = (__bf16)f.w;
    *(v4bf_t*)&As[0][row * 40 + c4] = b;
  }
  if (w == 0) __builtin_amdgcn_s_wait_tensorcnt(0);
  __syncthreads();

  v8f acc[2][4];
  #pragma unroll
  for (int fm = 0; fm < 2; ++fm)
    #pragma unroll
    for (int fn = 0; fn < 4; ++fn)
      #pragma unroll
      for (int i = 0; i < 8; ++i) acc[fm][fn][i] = 0.0f;

  for (int kt = 0; kt < 32; ++kt) {
    const int buf = kt & 1;
    const bool hasNext = (kt + 1) < 32;
    // kick off next B tile DMA into the other buffer (prev reads already drained
    // by the trailing barrier of the previous iteration)
    if (hasNext && w == 0) {
      tdm_load_2d_bf16((unsigned)(size_t)&Bs[buf ^ 1][0],
                       Wc + (size_t)tileN * 1024 + (kt + 1) * 32,
                       32, 128, 1024, 3, 3);
    }
    float4 aR[4];
    if (hasNext) {                      // prefetch next A K-stage into registers
      const int k0 = (kt + 1) * 32;
      #pragma unroll
      for (int it = 0; it < 4; ++it) {
        int v = tid + 256 * it; int row = v >> 3; int c4 = (v & 7) * 4;
        aR[it] = *(const float4*)&U[(size_t)(tileM + row) * 1024 + k0 + c4];
      }
    }
    v16bf a[2], b[4];
    #pragma unroll
    for (int fm = 0; fm < 2; ++fm) a[fm] = frag_a(As[buf], 40, waveM * 32 + fm * 16, lane);
    #pragma unroll
    for (int fn = 0; fn < 4; ++fn) b[fn] = frag_b(Bs[buf], 40, waveN * 64 + fn * 16, lane);
    #pragma unroll
    for (int fm = 0; fm < 2; ++fm)
      #pragma unroll
      for (int fn = 0; fn < 4; ++fn)
        acc[fm][fn] = wmma_bf16(a[fm], b[fn], acc[fm][fn]);
    if (hasNext) {
      const int nb = buf ^ 1;
      #pragma unroll
      for (int it = 0; it < 4; ++it) {
        int v = tid + 256 * it; int row = v >> 3; int c4 = (v & 7) * 4;
        v4bf_t bb; bb[0] = (__bf16)aR[it].x; bb[1] = (__bf16)aR[it].y;
        bb[2] = (__bf16)aR[it].z; bb[3] = (__bf16)aR[it].w;
        *(v4bf_t*)&As[nb][row * 40 + c4] = bb;
      }
      if (w == 0) __builtin_amdgcn_s_wait_tensorcnt(0);
    }
    __syncthreads();
  }

  // C/D layout: VGPR v = row M=v (lanes 0-15) / M=v+8 (lanes 16-31), N=lane&15
  const int lc = lane & 15, hv = lane >> 4;
  #pragma unroll
  for (int fm = 0; fm < 2; ++fm)
    #pragma unroll
    for (int fn = 0; fn < 4; ++fn)
      #pragma unroll
      for (int v = 0; v < 8; ++v) {
        int row = tileM + waveM * 32 + fm * 16 + v + hv * 8;
        int col = tileN + waveN * 64 + fn * 16 + lc;
        P[(size_t)row * 512 + col] = acc[fm][fn][v];
      }
}

// ---------------- nonlinearity / normalization (in place on P) ----------------
// Row layout (450 used of 512): [0,64)=A->S, [64,320)=K->Khat, [320,322)=beta, [322,450)=Bt

__global__ __launch_bounds__(256) void nonlin_kernel(
    float* __restrict__ P, const float* __restrict__ b_A, const float* __restrict__ b_k,
    const float* __restrict__ b_beta, const float* __restrict__ B_b) {
  const int wave = threadIdx.x >> 5, lane = threadIdx.x & 31;
  const int token = blockIdx.x * 8 + wave;         // 16384 tokens
  float* F = P + (size_t)token * 512;

  #pragma unroll
  for (int s = 0; s < 2; ++s) {                    // A -> S = 1/(1+DT^2 A)
    int m = s * 32 + lane;
    float a = F[m] + b_A[m];
    a = fminf(fmaxf(a, 0.0f), 100.0f);
    F[m] = 1.0f / (1.0f + 0.01f * a);
  }
  #pragma unroll
  for (int j = 0; j < 2; ++j) {                    // normalize K rows
    float v[4]; float ss = 0.0f;
    #pragma unroll
    for (int s = 0; s < 4; ++s) {
      int n = s * 32 + lane;
      v[s] = F[64 + j * 128 + n] + b_k[j * 128 + n];
      ss += v[s] * v[s];
    }
    #pragma unroll
    for (int o = 16; o > 0; o >>= 1) ss += __shfl_xor(ss, o, 32);
    float scale = 1.0f / fmaxf(sqrtf(ss), 1e-12f);
    #pragma unroll
    for (int s = 0; s < 4; ++s) F[64 + j * 128 + s * 32 + lane] = v[s] * scale;
  }
  if (lane < 2) {                                  // beta = 2*sigmoid
    float x = F[320 + lane] + b_beta[lane];
    F[320 + lane] = 2.0f / (1.0f + __expf(-x));
  }
  #pragma unroll
  for (int s = 0; s < 4; ++s) {                    // Bt += bias
    int n = s * 32 + lane;
    F[322 + n] += B_b[n];
  }
}

// ---------------- sequential scan: one wave per batch ----------------
// State n = s*32+lane; slots 0,1 = z (n<64), slots 2,3 = y (n>=64).

__global__ __launch_bounds__(32) void scan_kernel(const float* __restrict__ P, __bf16* __restrict__ hs) {
  const int b = blockIdx.x, lane = threadIdx.x;
  const float* F = P + (size_t)b * 2048 * 512;
  float h0 = 0.f, h1 = 0.f, h2 = 0.f, h3 = 0.f;
  for (int t = 0; t < 2048; ++t) {
    const float* R = F + (size_t)t * 512;
    if (t + 1 < 2048) __builtin_prefetch(R + 512 + lane * 16, 0, 0);  // global_prefetch_b8
    // oscillatory rotation: z' = S z - ((1-S)/DT) y ; y' = DT S z + S y
    float sa = R[lane], sb = R[lane + 32];
    float ca = (1.0f - sa) * 10.0f, cb = (1.0f - sb) * 10.0f;
    float z0 = h0, z1 = h1, y0 = h2, y1 = h3;
    h0 = sa * z0 - ca * y0;
    h1 = sb * z1 - cb * y1;
    h2 = 0.1f * sa * z0 + sa * y0;
    h3 = 0.1f * sb * z1 + sb * y1;
    // delta-product Householder updates
    #pragma unroll
    for (int j = 0; j < 2; ++j) {
      float beta = R[320 + j];
      float k0v = R[64 + j * 128 + lane];
      float k1v = R[64 + j * 128 + 32 + lane];
      float k2v = R[64 + j * 128 + 64 + lane];
      float k3v = R[64 + j * 128 + 96 + lane];
      float dp = k0v * h0 + k1v * h1 + k2v * h2 + k3v * h3;
      #pragma unroll
      for (int o = 16; o > 0; o >>= 1) dp += __shfl_xor(dp, o, 32);
      float bd = beta * dp;
      h0 -= bd * k0v; h1 -= bd * k1v; h2 -= bd * k2v; h3 -= bd * k3v;
    }
    h0 += R[322 + lane]; h1 += R[322 + 32 + lane];
    h2 += R[322 + 64 + lane]; h3 += R[322 + 96 + lane];
    __bf16* o = hs + ((size_t)b * 2048 + t) * 128;
    o[lane] = (__bf16)h0; o[32 + lane] = (__bf16)h1;
    o[64 + lane] = (__bf16)h2; o[96 + lane] = (__bf16)h3;
  }
}

// ---------------- GEMM2: out[16384x1024] = hs[16384x128] * Cb^T + u*D ----------------
// Both operands bf16 -> both tiles DMA'd by the TDM into padded LDS (stride 72:
// 64 data + 8 pad bf16 per row; pad_interval 32 dwords, pad_amount 4 dwords).
// K=128 processed as two K=64 halves, each one TDM pair.

__global__ __launch_bounds__(256) void gemm2_kernel(
    const __bf16* __restrict__ hsB, const __bf16* __restrict__ Cb,
    const float* __restrict__ U, const float* __restrict__ Dv, float* __restrict__ out) {
  __shared__ __bf16 As[128 * 72];
  __shared__ __bf16 Bs[128 * 72];
  const int tid = threadIdx.x, lane = tid & 31, w = tid >> 5;
  const int tileM = blockIdx.y * 128, tileN = blockIdx.x * 128;
  const int waveM = w & 3, waveN = w >> 2;

  v8f acc[2][4];
  #pragma unroll
  for (int fm = 0; fm < 2; ++fm)
    #pragma unroll
    for (int fn = 0; fn < 4; ++fn)
      #pragma unroll
      for (int i = 0; i < 8; ++i) acc[fm][fn][i] = 0.0f;

  #pragma unroll
  for (int half = 0; half < 2; ++half) {
    if (half) __syncthreads();                     // drain prior reads before overwrite
    if (w == 0) {
      const int k0 = half * 64;
      tdm_load_2d_bf16((unsigned)(size_t)&As[0], hsB + (size_t)tileM * 128 + k0,
                       /*tile_x=*/64, /*rows=*/128, /*stride=*/128,
                       /*pad_interval: 32 dw=*/4, /*pad_amount: 4 dw=*/3);
      tdm_load_2d_bf16((unsigned)(size_t)&Bs[0], Cb + (size_t)tileN * 128 + k0,
                       64, 128, 128, 4, 3);
      __builtin_amdgcn_s_wait_tensorcnt(0);
    }
    __syncthreads();
    #pragma unroll
    for (int ks = 0; ks < 2; ++ks) {
      const int kk = ks * 32;
      v16bf a[2], b[4];
      #pragma unroll
      for (int fm = 0; fm < 2; ++fm) a[fm] = frag_a(As + kk, 72, waveM * 32 + fm * 16, lane);
      #pragma unroll
      for (int fn = 0; fn < 4; ++fn) b[fn] = frag_b(Bs + kk, 72, waveN * 64 + fn * 16, lane);
      #pragma unroll
      for (int fm = 0; fm < 2; ++fm)
        #pragma unroll
        for (int fn = 0; fn < 4; ++fn)
          acc[fm][fn] = wmma_bf16(a[fm], b[fn], acc[fm][fn]);
    }
  }

  const int lc = lane & 15, hv = lane >> 4;
  #pragma unroll
  for (int fm = 0; fm < 2; ++fm)
    #pragma unroll
    for (int fn = 0; fn < 4; ++fn) {
      int col = tileN + waveN * 64 + fn * 16 + lc;
      float d = Dv[col];
      #pragma unroll
      for (int v = 0; v < 8; ++v) {
        int row = tileM + waveM * 32 + fm * 16 + v + hv * 8;
        size_t idx = (size_t)row * 1024 + col;
        out[idx] = acc[fm][fn][v] + U[idx] * d;
      }
    }
}

// ---------------- launch ----------------

extern "C" void kernel_launch(void* const* d_in, const int* in_sizes, int n_in,
                              void* d_out, int out_size, void* d_ws, size_t ws_size,
                              hipStream_t stream) {
  const float* u      = (const float*)d_in[0];
  const float* W_A    = (const float*)d_in[1];
  const float* b_A    = (const float*)d_in[2];
  const float* W_k    = (const float*)d_in[3];
  const float* b_k    = (const float*)d_in[4];
  const float* W_beta = (const float*)d_in[5];
  const float* b_beta = (const float*)d_in[6];
  const float* B_w    = (const float*)d_in[7];
  const float* B_b    = (const float*)d_in[8];
  const float* C      = (const float*)d_in[9];
  const float* Dv     = (const float*)d_in[10];
  float* out = (float*)d_out;

  char* ws = (char*)d_ws;
  __bf16* Wc  = (__bf16*)ws;                         // 512*1024*2   = 1 MiB
  __bf16* Cb  = (__bf16*)(ws + (1u << 20));          // 1024*128*2   = 256 KiB
  float*  P   = (float*)(ws + (1u << 20) + (256u << 10));       // 16384*512*4 = 32 MiB
  __bf16* hsB = (__bf16*)(ws + (1u << 20) + (256u << 10) + (32u << 20)); // 16384*128*2 = 4 MiB

  pack_w_kernel<<<2048, 256, 0, stream>>>(W_A, W_k, W_beta, B_w, Wc);
  pack_c_kernel<<<512, 256, 0, stream>>>(C, Cb);
  gemm1_kernel<<<dim3(4, 128), 256, 0, stream>>>(u, Wc, P);
  nonlin_kernel<<<2048, 256, 0, stream>>>(P, b_A, b_k, b_beta, B_b);
  scan_kernel<<<8, 32, 0, stream>>>(P, hsB);
  gemm2_kernel<<<dim3(8, 128), 256, 0, stream>>>(hsB, Cb, u, Dv, out);
}
